// TensorNet_33457795236213
// MI455X (gfx1250) — matrix operand order repaired and checked by reading, not compiled, not run on hardware
//
#include <hip/hip_runtime.h>

typedef __bf16 bf16;
typedef __attribute__((ext_vector_type(8)))  bf16  v8bf;
typedef __attribute__((ext_vector_type(16))) bf16  v16bf;
typedef __attribute__((ext_vector_type(8)))  float v8f;
typedef unsigned int v4ui __attribute__((ext_vector_type(4)));
typedef int v8i __attribute__((ext_vector_type(8)));
typedef int v4i __attribute__((ext_vector_type(4)));

#define N_NODES 5000
#define N_EDGES 80000
#define HCH 128
#define RBF 32
#define NLAY 2
#define NPAD 5008              // 313*16
#define RCUT 4.5f
#define PI_F 3.14159265358979f
#define KMAX 384               // largest K of any GEMM in this net

// ---------------------------------------------------------------- WMMA GEMM
// out[m,n] = epilogue( sum_k A[m,k]*W[n,k] + bias[n] )
// A: [Mpad,K] bf16 row-major (Mpad%16==0, K%32==0) streamed from global (HBM/L2)
// W: [Nout,K] bf16 row-major (weights). One 64xK panel is staged into LDS per
// workgroup by the Tensor Data Mover (single DMA, TENSORcnt-tracked) and then
// reused by all 8 waves (8 M-tiles x one N-tile per block).
static __device__ __forceinline__ v16bf join16(v8bf lo, v8bf hi) {
  v16bf r;
#pragma unroll
  for (int i = 0; i < 8; i++) { r[i] = lo[i]; r[i + 8] = hi[i]; }
  return r;
}

__global__ void __launch_bounds__(256) k_gemm(
    const bf16* __restrict__ A, const bf16* __restrict__ W,
    const float* __restrict__ bias, const float* __restrict__ rowscale,
    float* __restrict__ outF, bf16* __restrict__ outB,
    int Mpad, int Mstore, int K, int Nout, int act)
{
  __shared__ bf16 Bp[64 * KMAX];            // 48 KB of the 320 KB WGP LDS
  const int n0 = blockIdx.y * 64;
#if defined(__has_builtin) && __has_builtin(__builtin_amdgcn_tensor_load_to_lds)
  if ((threadIdx.x >> 5) == 0) {
    // 2-D tensor DMA descriptor (D#): tile = [rows=64][cols=K] bf16, row-major.
    unsigned long long ga = (unsigned long long)(uintptr_t)(W + (size_t)n0 * K);
    unsigned ldsa = (unsigned)(uintptr_t)(void*)&Bp[0];
    v4ui g0;
    g0[0] = 1u;                                            // count=1 (valid D#)
    g0[1] = ldsa;                                          // lds_addr
    g0[2] = (unsigned)(ga & 0xFFFFFFFFu);                  // global_addr[31:0]
    g0[3] = (unsigned)((ga >> 32) & 0x1FFFFFFu) | (2u << 30); // addr[56:32]|type=2
    v8i g1;
    g1[0] = (int)(1u << 16);                               // data_size=1 (2 bytes)
    g1[1] = (int)((unsigned)(K & 0xFFFF) << 16);           // tensor_dim0[15:0]
    g1[2] = (int)((unsigned)(Nout & 0xFFFF) << 16);        // tensor_dim1[15:0]
    g1[3] = (int)((unsigned)(K & 0xFFFF) << 16);           // tile_dim0 = K
    g1[4] = 64;                                            // tile_dim1 = 64 rows
    g1[5] = K;                                             // tensor_dim0_stride
    g1[6] = 0; g1[7] = 0;
    v4i gz; gz[0] = 0; gz[1] = 0; gz[2] = 0; gz[3] = 0;
    v8i gz8; gz8[0] = 0; gz8[1] = 0; gz8[2] = 0; gz8[3] = 0;
    gz8[4] = 0; gz8[5] = 0; gz8[6] = 0; gz8[7] = 0;
    __builtin_amdgcn_tensor_load_to_lds(g0, g1, gz, gz, gz8, 0);
    __builtin_amdgcn_s_wait_tensorcnt(0);
  }
#else
  for (int i = threadIdx.x; i < 64 * (K / 8); i += 256) {
    int nrow = i / (K / 8), kc = i % (K / 8);
    *(v8bf*)&Bp[(size_t)nrow * K + kc * 8] =
        *(const v8bf*)(W + (size_t)(n0 + nrow) * K + kc * 8);
  }
#endif
  __syncthreads();

  int tilesY = Mpad >> 4;
  int ty = blockIdx.x * 8 + (threadIdx.x >> 5);
  if (ty >= tilesY) return;                 // only after the barrier

  int lane = threadIdx.x & 31;
  int half = lane >> 4, l16 = lane & 15;
  // A fragment (ISA 16-bit A layout): lane row=ty*16+l16;
  //   e<8 -> k=e+half*8 ; e>=8 -> k=e+8+half*8  (two contiguous 8x bf16 chunks)
  const bf16* Arow = A + (size_t)(ty * 16 + l16) * K;
  // B fragment from LDS: lane col=l16 (+j*16), same k order as A (two 8x chunks)
  const bf16* Bl = Bp + (size_t)l16 * K;
  v8f acc[4] = {};
  for (int k0 = 0; k0 < K; k0 += 32) {
    __builtin_prefetch(Arow + k0 + 64, 0, 1);   // global_prefetch_b8
    v16bf av = join16(*(const v8bf*)(Arow + k0 + half * 8),
                      *(const v8bf*)(Arow + k0 + 16 + half * 8));
#pragma unroll
    for (int j = 0; j < 4; j++) {
      const bf16* bp = Bl + (size_t)j * 16 * K + k0 + half * 8;
      v16bf bv = join16(*(const v8bf*)bp, *(const v8bf*)(bp + 16));
      acc[j] = __builtin_amdgcn_wmma_f32_16x16x32_bf16(
          false, av, false, bv, (short)0, acc[j], false, false);
    }
  }
#pragma unroll
  for (int j = 0; j < 4; j++) {
    int n = n0 + j * 16 + l16;
    float bi = bias ? bias[n] : 0.f;
#pragma unroll
    for (int r = 0; r < 8; r++) {
      int m = ty * 16 + r + 8 * half;      // C/D layout: lanes>=16 -> M+8
      float v = acc[j][r] + bi;
      if (act) v = v / (1.f + __expf(-v)); // silu
      if (rowscale) v *= rowscale[m];
      if (m < Mstore) {
        size_t o = (size_t)m * Nout + n;
        if (outF) outF[o] = v;
        if (outB) outB[o] = (bf16)v;
      }
    }
  }
}

// ---------------------------------------------------------------- helpers
__global__ void k_f2b(const float* __restrict__ s, bf16* __restrict__ d, size_t n) {
  size_t t = (size_t)blockIdx.x * blockDim.x + threadIdx.x;
  if (t < n) d[t] = (bf16)s[t];
}

static __device__ __forceinline__ void atomic_addf(float* p, float v) {
  __hip_atomic_fetch_add(p, v, __ATOMIC_RELAXED, __HIP_MEMORY_SCOPE_AGENT);
}

// full 3x3 = ci*diag(i) + ca*skew(a) + cs*sym(s) + add*ones
static __device__ __forceinline__ void build_full(
    float i, float a0, float a1, float a2,
    float s00, float s11, float s22, float s01, float s02, float s12,
    float ci, float ca, float cs, float add, float* X)
{
  X[0] = ci * i + cs * s00 + add;   X[1] = -ca * a2 + cs * s01 + add;  X[2] = ca * a1 + cs * s02 + add;
  X[3] = ca * a2 + cs * s01 + add;  X[4] = ci * i + cs * s11 + add;    X[5] = -ca * a0 + cs * s12 + add;
  X[6] = -ca * a1 + cs * s02 + add; X[7] = ca * a0 + cs * s12 + add;   X[8] = ci * i + cs * s22 + add;
}

// ---------------------------------------------------------------- edge geometry + RBF
__global__ void k_edge_geom(const float* __restrict__ ev, const int* __restrict__ src,
                            const int* __restrict__ dst, const float* __restrict__ means,
                            const float* __restrict__ betas, float* __restrict__ Cc,
                            float* __restrict__ vh, bf16* __restrict__ eattr)
{
  int e = blockIdx.x * blockDim.x + threadIdx.x;
  if (e >= N_EDGES) return;
  float x = ev[3 * e], y = ev[3 * e + 1], z = ev[3 * e + 2];
  float d = sqrtf(x * x + y * y + z * z);
  float C = (d < RCUT) ? 0.5f * (__cosf(d * PI_F / RCUT) + 1.f) : 0.f;
  Cc[e] = C;
  bool deg = (src[e] == dst[e]) || !(d > 0.f);
  float inv = deg ? 1.f : 1.f / d;
  vh[3 * e] = x * inv; vh[3 * e + 1] = y * inv; vh[3 * e + 2] = z * inv;
  float ed = __expf(-(5.f / RCUT) * d);
  for (int r = 0; r < RBF; r++) {
    float t = ed - means[r];
    eattr[(size_t)e * RBF + r] = (bf16)(C * __expf(-betas[r] * t * t));
  }
}

// gather concat(Z[src],Z[dst]) -> bf16 [E,256]
__global__ void k_gatherZ(const int* __restrict__ z, const int* __restrict__ src,
                          const int* __restrict__ dst, const float* __restrict__ emb,
                          bf16* __restrict__ Zab)
{
  size_t t = (size_t)blockIdx.x * blockDim.x + threadIdx.x;
  if (t >= (size_t)N_EDGES * 256) return;
  int e = (int)(t >> 8), c = (int)(t & 255);
  int node = (c < HCH) ? src[e] : dst[e];
  Zab[t] = (bf16)emb[(size_t)z[node] * HCH + (c & (HCH - 1))];
}

// scatter the per-edge tensors into compact node components (10 planes)
__global__ void k_scatter(const float* __restrict__ Wcat, const float* __restrict__ wE,
                          const float* __restrict__ vh, const int* __restrict__ src,
                          float* __restrict__ comps)
{
  size_t t = (size_t)blockIdx.x * blockDim.x + threadIdx.x;
  if (t >= (size_t)N_EDGES * HCH) return;
  int e = (int)(t >> 7), h = (int)(t & 127);
  float w = wE[(size_t)e * HCH + h];
  const float* wc = Wcat + (size_t)e * 384;
  float c1 = wc[h] * w, c2 = wc[128 + h] * w, c3 = wc[256 + h] * w;
  float v0 = vh[3 * e], v1 = vh[3 * e + 1], v2 = vh[3 * e + 2];
  float q = (v0 * v0 + v1 * v1 + v2 * v2) * (1.f / 3.f);
  size_t nd = (size_t)src[e] * HCH + h;
  const size_t P = (size_t)NPAD * HCH;
  float vals[10] = { c1, c2 * v0, c2 * v1, c2 * v2,
                     c3 * (v0 * v0 - q), c3 * (v1 * v1 - q), c3 * (v2 * v2 - q),
                     c3 * v0 * v1, c3 * v0 * v2, c3 * v1 * v2 };
#pragma unroll
  for (int p = 0; p < 10; p++) atomic_addf(comps + p * P + nd, vals[p]);
}

// squared Frobenius norm of compact I+A+S (cross terms vanish)
__global__ void k_tnorm_comps(const float* __restrict__ c, float* __restrict__ tn)
{
  size_t t = (size_t)blockIdx.x * blockDim.x + threadIdx.x;
  const size_t P = (size_t)NPAD * HCH;
  if (t >= P) return;
  float i = c[t], a0 = c[P + t], a1 = c[2 * P + t], a2 = c[3 * P + t];
  float s00 = c[4 * P + t], s11 = c[5 * P + t], s22 = c[6 * P + t];
  float s01 = c[7 * P + t], s02 = c[8 * P + t], s12 = c[9 * P + t];
  tn[t] = 3.f * i * i + 2.f * (a0 * a0 + a1 * a1 + a2 * a2)
        + s00 * s00 + s11 * s11 + s22 * s22 + 2.f * (s01 * s01 + s02 * s02 + s12 * s12);
}

__global__ void k_layernorm(float* __restrict__ x, const float* __restrict__ g,
                            const float* __restrict__ b, int D)
{
  __shared__ float red[128];
  int n = blockIdx.x;
  float* row = x + (size_t)n * D;
  float s = 0.f;
  for (int i = threadIdx.x; i < D; i += 128) s += row[i];
  red[threadIdx.x] = s; __syncthreads();
  for (int o = 64; o > 0; o >>= 1) { if (threadIdx.x < o) red[threadIdx.x] += red[threadIdx.x + o]; __syncthreads(); }
  float mean = red[0] / D; __syncthreads();
  float v = 0.f;
  for (int i = threadIdx.x; i < D; i += 128) { float d = row[i] - mean; v += d * d; }
  red[threadIdx.x] = v; __syncthreads();
  for (int o = 64; o > 0; o >>= 1) { if (threadIdx.x < o) red[threadIdx.x] += red[threadIdx.x + o]; __syncthreads(); }
  float inv = rsqrtf(red[0] / D + 1e-5f);
  __syncthreads();
  for (int i = threadIdx.x; i < D; i += 128) row[i] = (row[i] - mean) * inv * g[i] + b[i];
}

// X = nrm0*I' + nrm1*A' + nrm2*S'   (te chan_mix has no bias)
__global__ void k_combine_te(const float* __restrict__ mix, const float* __restrict__ nrmv,
                             float* __restrict__ X)
{
  size_t t = (size_t)blockIdx.x * blockDim.x + threadIdx.x;
  const size_t P = (size_t)NPAD * HCH;
  if (t >= P) return;
  size_t n = t >> 7; int h = (int)(t & 127);
  float n0 = nrmv[n * 384 + h * 3], n1 = nrmv[n * 384 + h * 3 + 1], n2 = nrmv[n * 384 + h * 3 + 2];
  float Xl[9];
  build_full(mix[t], mix[P + t], mix[2 * P + t], mix[3 * P + t],
             mix[4 * P + t], mix[5 * P + t], mix[6 * P + t],
             mix[7 * P + t], mix[8 * P + t], mix[9 * P + t], n0, n1, n2, 0.f, Xl);
#pragma unroll
  for (int j = 0; j < 9; j++) X[t * 9 + j] = Xl[j];
}

// X <- X/(tnorm+1) ; decompose X into compact comps
__global__ void k_normdecomp(float* __restrict__ X, float* __restrict__ c)
{
  size_t t = (size_t)blockIdx.x * blockDim.x + threadIdx.x;
  const size_t P = (size_t)NPAD * HCH;
  if (t >= P) return;
  float Xl[9]; float ss = 0.f;
#pragma unroll
  for (int j = 0; j < 9; j++) { Xl[j] = X[t * 9 + j]; ss += Xl[j] * Xl[j]; }
  float inv = 1.f / (ss + 1.f);
#pragma unroll
  for (int j = 0; j < 9; j++) { Xl[j] *= inv; X[t * 9 + j] = Xl[j]; }
  float iso = (Xl[0] + Xl[4] + Xl[8]) * (1.f / 3.f);
  c[t] = iso;
  c[P + t]     = 0.5f * (Xl[7] - Xl[5]);
  c[2 * P + t] = 0.5f * (Xl[2] - Xl[6]);
  c[3 * P + t] = 0.5f * (Xl[3] - Xl[1]);
  c[4 * P + t] = Xl[0] - iso;
  c[5 * P + t] = Xl[4] - iso;
  c[6 * P + t] = Xl[8] - iso;
  c[7 * P + t] = 0.5f * (Xl[1] + Xl[3]);
  c[8 * P + t] = 0.5f * (Xl[2] + Xl[6]);
  c[9 * P + t] = 0.5f * (Xl[5] + Xl[7]);
}

// Y = I1+A1+S1 (mixed comps + sum of the three il1 biases on every entry)
__global__ void k_buildY(const float* __restrict__ mix, const float* __restrict__ b3,
                         float* __restrict__ Y)
{
  size_t t = (size_t)blockIdx.x * blockDim.x + threadIdx.x;
  const size_t P = (size_t)NPAD * HCH;
  if (t >= P) return;
  int h = (int)(t & 127);
  float add = b3[h] + b3[HCH + h] + b3[2 * HCH + h];
  float Yl[9];
  build_full(mix[t], mix[P + t], mix[2 * P + t], mix[3 * P + t],
             mix[4 * P + t], mix[5 * P + t], mix[6 * P + t],
             mix[7 * P + t], mix[8 * P + t], mix[9 * P + t], 1.f, 1.f, 1.f, add, Yl);
#pragma unroll
  for (int j = 0; j < 9; j++) Y[t * 9 + j] = Yl[j];
}

// messages: msg[src] += ea0*I1[dst] + ea1*A1[dst] + ea2*S1[dst]
__global__ void k_msg(const float* __restrict__ mix, const float* __restrict__ b3,
                      const float* __restrict__ ea, const int* __restrict__ src,
                      const int* __restrict__ dst, float* __restrict__ msg)
{
  size_t t = (size_t)blockIdx.x * blockDim.x + threadIdx.x;
  if (t >= (size_t)N_EDGES * HCH) return;
  int e = (int)(t >> 7), h = (int)(t & 127);
  const size_t P = (size_t)NPAD * HCH;
  size_t nd = (size_t)dst[e] * HCH + h;
  float e0 = ea[(size_t)e * 384 + h * 3];
  float e1 = ea[(size_t)e * 384 + h * 3 + 1];
  float e2 = ea[(size_t)e * 384 + h * 3 + 2];
  float add = e0 * b3[h] + e1 * b3[HCH + h] + e2 * b3[2 * HCH + h];
  float Ml[9];
  build_full(mix[nd], mix[P + nd], mix[2 * P + nd], mix[3 * P + nd],
             mix[4 * P + nd], mix[5 * P + nd], mix[6 * P + nd],
             mix[7 * P + nd], mix[8 * P + nd], mix[9 * P + nd], e0, e1, e2, add, Ml);
  float* base = msg + ((size_t)src[e] * HCH + h) * 9;
#pragma unroll
  for (int j = 0; j < 9; j++) atomic_addf(base + j, Ml[j]);
}

// P = msg@Y + Y@msg ; comps = decompose(P)/(tnorm(P)+1)
__global__ void k_py(const float* __restrict__ msg, const float* __restrict__ Y,
                     float* __restrict__ c)
{
  size_t t = (size_t)blockIdx.x * blockDim.x + threadIdx.x;
  const size_t P1 = (size_t)NPAD * HCH;
  if (t >= P1) return;
  float M[9], Yl[9], Pm[9];
#pragma unroll
  for (int j = 0; j < 9; j++) { M[j] = msg[t * 9 + j]; Yl[j] = Y[t * 9 + j]; }
  float ss = 0.f;
#pragma unroll
  for (int r = 0; r < 3; r++)
#pragma unroll
    for (int cc = 0; cc < 3; cc++) {
      float acc = 0.f;
#pragma unroll
      for (int k = 0; k < 3; k++) acc += M[r * 3 + k] * Yl[k * 3 + cc] + Yl[r * 3 + k] * M[k * 3 + cc];
      Pm[r * 3 + cc] = acc; ss += acc * acc;
    }
  float inv = 1.f / (ss + 1.f);
  float iso = (Pm[0] + Pm[4] + Pm[8]) * (1.f / 3.f);
  c[t]            = iso * inv;
  c[P1 + t]       = 0.5f * (Pm[7] - Pm[5]) * inv;
  c[2 * P1 + t]   = 0.5f * (Pm[2] - Pm[6]) * inv;
  c[3 * P1 + t]   = 0.5f * (Pm[3] - Pm[1]) * inv;
  c[4 * P1 + t]   = (Pm[0] - iso) * inv;
  c[5 * P1 + t]   = (Pm[4] - iso) * inv;
  c[6 * P1 + t]   = (Pm[8] - iso) * inv;
  c[7 * P1 + t]   = 0.5f * (Pm[1] + Pm[3]) * inv;
  c[8 * P1 + t]   = 0.5f * (Pm[2] + Pm[6]) * inv;
  c[9 * P1 + t]   = 0.5f * (Pm[5] + Pm[7]) * inv;
}

// X += dX + dX@dX
__global__ void k_updatex(const float* __restrict__ mix, const float* __restrict__ b3,
                          float* __restrict__ X)
{
  size_t t = (size_t)blockIdx.x * blockDim.x + threadIdx.x;
  const size_t P = (size_t)NPAD * HCH;
  if (t >= P) return;
  int h = (int)(t & 127);
  float add = b3[h] + b3[HCH + h] + b3[2 * HCH + h];
  float D[9];
  build_full(mix[t], mix[P + t], mix[2 * P + t], mix[3 * P + t],
             mix[4 * P + t], mix[5 * P + t], mix[6 * P + t],
             mix[7 * P + t], mix[8 * P + t], mix[9 * P + t], 1.f, 1.f, 1.f, add, D);
  float Xl[9];
#pragma unroll
  for (int j = 0; j < 9; j++) Xl[j] = X[t * 9 + j];
#pragma unroll
  for (int r = 0; r < 3; r++)
#pragma unroll
    for (int cc = 0; cc < 3; cc++) {
      float acc = D[r * 3 + cc];
#pragma unroll
      for (int k = 0; k < 3; k++) acc += D[r * 3 + k] * D[k * 3 + cc];
      Xl[r * 3 + cc] += acc;
    }
#pragma unroll
  for (int j = 0; j < 9; j++) X[t * 9 + j] = Xl[j];
}

// head: x3h = concat(tnorm(I),tnorm(A),tnorm(S))
__global__ void k_head(const float* __restrict__ X, float* __restrict__ x3h)
{
  size_t t = (size_t)blockIdx.x * blockDim.x + threadIdx.x;
  const size_t P = (size_t)NPAD * HCH;
  if (t >= P) return;
  float Xl[9];
#pragma unroll
  for (int j = 0; j < 9; j++) Xl[j] = X[t * 9 + j];
  float iso = (Xl[0] + Xl[4] + Xl[8]) * (1.f / 3.f);
  float v0 = 0.5f * (Xl[7] - Xl[5]), v1 = 0.5f * (Xl[2] - Xl[6]), v2 = 0.5f * (Xl[3] - Xl[1]);
  float s00 = Xl[0] - iso, s11 = Xl[4] - iso, s22 = Xl[8] - iso;
  float s01 = 0.5f * (Xl[1] + Xl[3]), s02 = 0.5f * (Xl[2] + Xl[6]), s12 = 0.5f * (Xl[5] + Xl[7]);
  size_t n = t >> 7; int h = (int)(t & 127);
  x3h[n * 384 + h]       = 3.f * iso * iso;
  x3h[n * 384 + 128 + h] = 2.f * (v0 * v0 + v1 * v1 + v2 * v2);
  x3h[n * 384 + 256 + h] = s00 * s00 + s11 * s11 + s22 * s22 + 2.f * (s01 * s01 + s02 * s02 + s12 * s12);
}

// ---------------------------------------------------------------- host
static inline void gemm(hipStream_t st, const bf16* A, const bf16* W, const float* bias,
                        const float* rs, float* oF, bf16* oB,
                        int Mpad, int Mstore, int K, int Nout, int act)
{
  int tilesY = Mpad / 16;
  dim3 grid((unsigned)((tilesY + 7) / 8), (unsigned)(Nout / 64));
  k_gemm<<<grid, dim3(256), 0, st>>>(A, W, bias, rs, oF, oB, Mpad, Mstore, K, Nout, act);
}

#define EW(count) dim3((unsigned)((((size_t)(count)) + 255) / 256)), dim3(256), 0, stream

extern "C" void kernel_launch(void* const* d_in, const int* in_sizes, int n_in,
                              void* d_out, int out_size, void* d_ws, size_t ws_size,
                              hipStream_t stream)
{
  (void)in_sizes; (void)n_in; (void)out_size; (void)ws_size;
  const int* z        = (const int*)d_in[0];
  const int* ei       = (const int*)d_in[1];
  const int* src      = ei;
  const int* dstp     = ei + N_EDGES;
  const float* evec   = (const float*)d_in[2];
  const float* means  = (const float*)d_in[3];
  const float* betas  = (const float*)d_in[4];
  const float* dp_w   = (const float*)d_in[5];
  const float* dp_b   = (const float*)d_in[6];
  const float* emb    = (const float*)d_in[7];
  const float* emb2_w = (const float*)d_in[8];
  const float* emb2_b = (const float*)d_in[9];
  const float* telt_w = (const float*)d_in[10];
  const float* tels1_w= (const float*)d_in[11];
  const float* tels1_b= (const float*)d_in[12];
  const float* tels2_w= (const float*)d_in[13];
  const float* tels2_b= (const float*)d_in[14];
  const float* teln_g = (const float*)d_in[15];
  const float* teln_b = (const float*)d_in[16];
  const float* il1_w  = (const float*)d_in[17];
  const float* il1_b  = (const float*)d_in[18];
  const float* i2a_w  = (const float*)d_in[19];
  const float* i2a_b  = (const float*)d_in[20];
  const float* i2b_w  = (const float*)d_in[21];
  const float* i2b_b  = (const float*)d_in[22];
  const float* i2c_w  = (const float*)d_in[23];
  const float* i2c_b  = (const float*)d_in[24];
  const float* il3_w  = (const float*)d_in[25];
  const float* il3_b  = (const float*)d_in[26];
  const float* on_g   = (const float*)d_in[27];
  const float* on_b   = (const float*)d_in[28];
  const float* out_w  = (const float*)d_in[29];
  const float* out_b  = (const float*)d_in[30];

  char* ws = (char*)d_ws;
  size_t off = 0;
  auto alloc = [&](size_t bytes) { size_t o = off; off = (off + bytes + 255) & ~(size_t)255; return o; };
  const size_t P1 = (size_t)NPAD * HCH;

  size_t oCc    = alloc((size_t)N_EDGES * 4);
  size_t oVh    = alloc((size_t)N_EDGES * 3 * 4);
  size_t oEat   = alloc((size_t)N_EDGES * RBF * 2);
  size_t oZab   = alloc((size_t)N_EDGES * 256 * 2);   // later reused as t1 (E x 128 bf16)
  size_t oWcat  = alloc((size_t)N_EDGES * 384 * 4);   // later reused as ea (E x 384 f32)
  size_t oWE    = alloc((size_t)N_EDGES * 128 * 4);   // later reused as t2 (E x 256 bf16)
  size_t oComps = alloc(10 * P1 * 4);
  size_t oCompsB= alloc(10 * P1 * 2);
  size_t oMix   = alloc(10 * P1 * 4);
  size_t oX     = alloc(P1 * 9 * 4);
  size_t oY     = alloc(P1 * 9 * 4);
  size_t oMsg   = alloc(P1 * 9 * 4);
  size_t oNrm   = alloc(P1 * 4);
  size_t oNrmB  = alloc(P1 * 2);
  size_t oH1B   = alloc((size_t)NPAD * 256 * 2);
  size_t oNrmv  = alloc((size_t)NPAD * 384 * 4);
  size_t oX3h   = alloc((size_t)NPAD * 384 * 4);
  size_t oX3hB  = alloc((size_t)NPAD * 384 * 2);
  size_t oWpool = alloc(2u * 1024 * 1024);

  float* Cc     = (float*)(ws + oCc);
  float* vh     = (float*)(ws + oVh);
  bf16*  eattr  = (bf16*)(ws + oEat);
  bf16*  Zab    = (bf16*)(ws + oZab);
  float* Wcat   = (float*)(ws + oWcat);
  float* wE     = (float*)(ws + oWE);
  float* comps  = (float*)(ws + oComps);
  bf16*  compsB = (bf16*)(ws + oCompsB);
  float* mix    = (float*)(ws + oMix);
  float* X      = (float*)(ws + oX);
  float* Y      = (float*)(ws + oY);
  float* msg    = (float*)(ws + oMsg);
  float* nrm    = (float*)(ws + oNrm);
  bf16*  nrmB   = (bf16*)(ws + oNrmB);
  bf16*  h1B    = (bf16*)(ws + oH1B);
  float* nrmv   = (float*)(ws + oNrmv);
  float* x3h    = (float*)(ws + oX3h);
  bf16*  x3hB   = (bf16*)(ws + oX3hB);
  bf16*  t1     = (bf16*)(ws + oZab);
  bf16*  t2     = (bf16*)(ws + oWE);
  float* ea     = (float*)(ws + oWcat);

  // bf16 weight pool layout
  bf16* wp = (bf16*)(ws + oWpool);
  bf16* dpWb   = wp;               size_t cur = 3 * 128 * 32;
  bf16* emb2Wb = wp + cur;         cur += 128 * 256;
  bf16* teltWb = wp + cur;         cur += 3 * 128 * 128;
  bf16* ls1Wb  = wp + cur;         cur += 256 * 128;
  bf16* ls2Wb  = wp + cur;         cur += 384 * 256;
  bf16* il1Wb  = wp + cur;         cur += (size_t)NLAY * 3 * 128 * 128;
  bf16* i2aWb  = wp + cur;         cur += (size_t)NLAY * 128 * 32;
  bf16* i2bWb  = wp + cur;         cur += (size_t)NLAY * 256 * 128;
  bf16* i2cWb  = wp + cur;         cur += (size_t)NLAY * 384 * 256;
  bf16* il3Wb  = wp + cur;         cur += (size_t)NLAY * 3 * 128 * 128;
  bf16* outWb  = wp + cur;         cur += 128 * 384;

  // convert all weights once per call
  k_f2b<<<EW(3 * 128 * 32)>>>(dp_w, dpWb, 3 * 128 * 32);
  k_f2b<<<EW(128 * 256)>>>(emb2_w, emb2Wb, 128 * 256);
  k_f2b<<<EW(3 * 128 * 128)>>>(telt_w, teltWb, 3 * 128 * 128);
  k_f2b<<<EW(256 * 128)>>>(tels1_w, ls1Wb, 256 * 128);
  k_f2b<<<EW(384 * 256)>>>(tels2_w, ls2Wb, 384 * 256);
  k_f2b<<<EW((size_t)NLAY * 3 * 128 * 128)>>>(il1_w, il1Wb, (size_t)NLAY * 3 * 128 * 128);
  k_f2b<<<EW((size_t)NLAY * 128 * 32)>>>(i2a_w, i2aWb, (size_t)NLAY * 128 * 32);
  k_f2b<<<EW((size_t)NLAY * 256 * 128)>>>(i2b_w, i2bWb, (size_t)NLAY * 256 * 128);
  k_f2b<<<EW((size_t)NLAY * 384 * 256)>>>(i2c_w, i2cWb, (size_t)NLAY * 384 * 256);
  k_f2b<<<EW((size_t)NLAY * 3 * 128 * 128)>>>(il3_w, il3Wb, (size_t)NLAY * 3 * 128 * 128);
  k_f2b<<<EW(128 * 384)>>>(out_w, outWb, 128 * 384);

  // ---- edge features ----
  k_edge_geom<<<EW(N_EDGES)>>>(evec, src, dstp, means, betas, Cc, vh, eattr);
  k_gatherZ<<<EW((size_t)N_EDGES * 256)>>>(z, src, dstp, emb, Zab);

  // W1|W2|W3 = edge_attr @ dp_w^T  (concatenated: Nout=384)
  gemm(stream, eattr, dpWb, dp_b, nullptr, Wcat, nullptr, N_EDGES, N_EDGES, 32, 384, 0);
  // w = C * (concat(Z)@emb2^T + b)
  gemm(stream, Zab, emb2Wb, emb2_b, Cc, wE, nullptr, N_EDGES, N_EDGES, 256, 128, 0);

  // ---- TensorEmbedding aggregation ----
  (void)hipMemsetAsync(comps, 0, 10 * P1 * 4, stream);
  k_scatter<<<EW((size_t)N_EDGES * HCH)>>>(Wcat, wE, vh, src, comps);
  k_tnorm_comps<<<EW(P1)>>>(comps, nrm);
  k_layernorm<<<dim3(NPAD), dim3(128), 0, stream>>>(nrm, teln_g, teln_b, 128);
  k_f2b<<<EW(P1)>>>(nrm, nrmB, P1);
  gemm(stream, nrmB, ls1Wb, tels1_b, nullptr, nullptr, h1B, NPAD, NPAD, 128, 256, 1);
  gemm(stream, h1B, ls2Wb, tels2_b, nullptr, nrmv, nullptr, NPAD, NPAD, 256, 384, 1);
  // te chan_mix (no bias): iso plane (M=NPAD), a planes (M=3*NPAD), s planes (M=6*NPAD)
  k_f2b<<<EW(10 * P1)>>>(comps, compsB, 10 * P1);
  gemm(stream, compsB,           teltWb,             nullptr, nullptr, mix,           nullptr, NPAD,   NPAD,   128, 128, 0);
  gemm(stream, compsB + P1,      teltWb + 128 * 128, nullptr, nullptr, mix + P1,      nullptr, 3*NPAD, 3*NPAD, 128, 128, 0);
  gemm(stream, compsB + 4 * P1,  teltWb + 2*128*128, nullptr, nullptr, mix + 4 * P1,  nullptr, 6*NPAD, 6*NPAD, 128, 128, 0);
  k_combine_te<<<EW(P1)>>>(mix, nrmv, X);

  // ---- interaction layers ----
  for (int l = 0; l < NLAY; l++) {
    const bf16* wa = i2aWb + (size_t)l * 128 * 32;
    const bf16* wb = i2bWb + (size_t)l * 256 * 128;
    const bf16* wc = i2cWb + (size_t)l * 384 * 256;
    const bf16* w1 = il1Wb + (size_t)l * 3 * 128 * 128;
    const bf16* w3 = il3Wb + (size_t)l * 3 * 128 * 128;
    const float* b1 = il1_b + (size_t)l * 3 * 128;
    const float* b3 = il3_b + (size_t)l * 3 * 128;
    // edge MLP chain (silu fused; last one scaled by C)
    gemm(stream, eattr, wa, i2a_b + (size_t)l * 128, nullptr, nullptr, t1, N_EDGES, N_EDGES, 32, 128, 1);
    gemm(stream, t1, wb, i2b_b + (size_t)l * 256, nullptr, nullptr, t2, N_EDGES, N_EDGES, 128, 256, 1);
    gemm(stream, t2, wc, i2c_b + (size_t)l * 384, Cc, ea, nullptr, N_EDGES, N_EDGES, 256, 384, 1);
    // node side
    k_normdecomp<<<EW(P1)>>>(X, comps);
    k_f2b<<<EW(10 * P1)>>>(comps, compsB, 10 * P1);
    gemm(stream, compsB,          w1,               nullptr, nullptr, mix,          nullptr, NPAD,   NPAD,   128, 128, 0);
    gemm(stream, compsB + P1,     w1 + 128 * 128,   nullptr, nullptr, mix + P1,     nullptr, 3*NPAD, 3*NPAD, 128, 128, 0);
    gemm(stream, compsB + 4 * P1, w1 + 2 * 128*128, nullptr, nullptr, mix + 4 * P1, nullptr, 6*NPAD, 6*NPAD, 128, 128, 0);
    k_buildY<<<EW(P1)>>>(mix, b1, Y);
    (void)hipMemsetAsync(msg, 0, P1 * 9 * 4, stream);
    k_msg<<<EW((size_t)N_EDGES * HCH)>>>(mix, b1, ea, src, dstp, msg);
    k_py<<<EW(P1)>>>(msg, Y, comps);
    k_f2b<<<EW(10 * P1)>>>(comps, compsB, 10 * P1);
    gemm(stream, compsB,          w3,               nullptr, nullptr, mix,          nullptr, NPAD,   NPAD,   128, 128, 0);
    gemm(stream, compsB + P1,     w3 + 128 * 128,   nullptr, nullptr, mix + P1,     nullptr, 3*NPAD, 3*NPAD, 128, 128, 0);
    gemm(stream, compsB + 4 * P1, w3 + 2 * 128*128, nullptr, nullptr, mix + 4 * P1, nullptr, 6*NPAD, 6*NPAD, 128, 128, 0);
    k_updatex<<<EW(P1)>>>(mix, b3, X);
  }

  // ---- output head ----
  k_head<<<EW(P1)>>>(X, x3h);
  k_layernorm<<<dim3(NPAD), dim3(128), 0, stream>>>(x3h, on_g, on_b, 384);
  k_f2b<<<EW((size_t)NPAD * 384)>>>(x3h, x3hB, (size_t)NPAD * 384);
  gemm(stream, x3hB, outWb, out_b, nullptr, (float*)d_out, nullptr, NPAD, N_NODES, 384, 128, 1);
}
